// Decoder_56143812494202
// MI455X (gfx1250) — matrix operand order repaired and checked by reading, not compiled, not run on hardware
//
#include <hip/hip_runtime.h>
#include <hip/hip_bf16.h>

// Problem constants from the reference (max_len is a device scalar; the
// reference fixes it to 30, and reading it host-side would break capture).
#define BB 128
#define II 1024
#define HH 1024
#define VV 50257
#define TT 30

typedef __attribute__((ext_vector_type(16))) __bf16 v16bf;
typedef __attribute__((ext_vector_type(8)))  __bf16 v8bf;
typedef __attribute__((ext_vector_type(8)))  float  v8f;

union ABFrag { v16bf v; v8bf h[2]; };

// ---------------------------------------------------------------- converts
__global__ void cvt_f32_bf16(const float* __restrict__ src,
                             __bf16* __restrict__ dst, long long n) {
    long long i = (long long)blockIdx.x * blockDim.x + threadIdx.x;
    long long stride = (long long)gridDim.x * blockDim.x;
    for (; i < n; i += stride) dst[i] = (__bf16)src[i];
}

// x0 = C*linC_w + linC_b  (bf16), and h0 = init_hidden (f32 + bf16 copies)
__global__ void init_kernel(const float* __restrict__ C,
                            const float* __restrict__ linC_w,
                            const float* __restrict__ linC_b,
                            const float* __restrict__ init_hidden,
                            __bf16* __restrict__ x0_bf,
                            float* __restrict__ h_f32,
                            __bf16* __restrict__ h_bf) {
    int idx = blockIdx.x * blockDim.x + threadIdx.x;   // B*I threads
    int b = idx >> 10, i = idx & 1023;
    float x0 = C[b] * linC_w[i] + linC_b[i];
    x0_bf[idx] = (__bf16)x0;
    float h = init_hidden[idx];
    h_f32[idx] = h;
    h_bf[idx]  = (__bf16)h;
}

// ------------------------------------------------- fused GRU step (WMMA)
// One wave computes a 16(batch) x 32(hidden) patch of h_new. It accumulates
// the six gate tiles (r/z/n for both x@W_ih^T and h@W_hh^T) with 12 WMMAs
// per K-step sharing two A-fragments, then applies the GRU nonlinearity and
// writes h_new (f32 + bf16). Reads h_prev, writes h_next (ping-pong buffers
// -- no grid sync needed). Gate/K offsets fold into load immediates.
__global__ __launch_bounds__(128) void
gru_step_fused(const __bf16* __restrict__ X, int ldx,    // [B, I] bf16
               const __bf16* __restrict__ Wih,           // [3H, I] bf16
               const __bf16* __restrict__ Whh,           // [3H, H] bf16
               const float*  __restrict__ b_ih,          // [3H]
               const float*  __restrict__ b_hh,          // [3H]
               const float*  __restrict__ h_prev,        // [B, H] f32
               const __bf16* __restrict__ h_prev_bf,     // [B, H] bf16
               float*  __restrict__ h_next,              // [B, H] f32
               __bf16* __restrict__ h_next_bf)           // [B, H] bf16
{
    const int wave   = blockIdx.x * (blockDim.x >> 5) + (threadIdx.x >> 5);
    const int mtiles = BB / 16;                       // 8
    if (wave >= mtiles * (HH / 32)) return;           // 256 waves total
    const int mtile = wave % mtiles;
    const int jbase = (wave / mtiles) << 5;           // 32-wide hidden strip

    const int lane = threadIdx.x & 31;
    const int l15  = lane & 15;
    const int akb  = (lane & 16) ? 16 : 0;
    const int bkb  = (lane & 16) ? 16 : 0;

    const __bf16* xptr = X + (size_t)(mtile * 16 + l15) * ldx + akb;
    const __bf16* hptr = h_prev_bf + (size_t)(mtile * 16 + l15) * HH + akb;

    // per-subtile weight-row base pointers; gate offset g*HH*II (2MB) and k0
    // are compile-time constants folded into the 24-bit instruction offset
    const __bf16* wih_b[2];
    const __bf16* whh_b[2];
    #pragma unroll
    for (int s = 0; s < 2; ++s) {
        const int col = jbase + s * 16 + l15;
        wih_b[s] = Wih + (size_t)col * II + bkb;
        whh_b[s] = Whh + (size_t)col * HH + bkb;
    }

    v8f gi[2][3], gh[2][3];
    #pragma unroll
    for (int s = 0; s < 2; ++s)
        #pragma unroll
        for (int g = 0; g < 3; ++g) { gi[s][g] = (v8f){}; gh[s][g] = (v8f){}; }

    for (int k0 = 0; k0 < II; k0 += 32) {
        ABFrag ax, ah;
        ax.h[0] = *reinterpret_cast<const v8bf*>(xptr + k0);
        ax.h[1] = *reinterpret_cast<const v8bf*>(xptr + k0 + 8);
        ah.h[0] = *reinterpret_cast<const v8bf*>(hptr + k0);
        ah.h[1] = *reinterpret_cast<const v8bf*>(hptr + k0 + 8);
        #pragma unroll
        for (int s = 0; s < 2; ++s) {
            #pragma unroll
            for (int g = 0; g < 3; ++g) {
                const size_t go_i = (size_t)g * HH * II;   // W_ih gate offset
                const size_t go_h = (size_t)g * HH * HH;   // W_hh gate offset
                ABFrag bi, bh;
                bi.h[0] = *reinterpret_cast<const v8bf*>(wih_b[s] + go_i + k0);
                bi.h[1] = *reinterpret_cast<const v8bf*>(wih_b[s] + go_i + k0 + 8);
                gi[s][g] = __builtin_amdgcn_wmma_f32_16x16x32_bf16(
                    false, ax.v, false, bi.v, (short)0, gi[s][g], false, false);
                bh.h[0] = *reinterpret_cast<const v8bf*>(whh_b[s] + go_h + k0);
                bh.h[1] = *reinterpret_cast<const v8bf*>(whh_b[s] + go_h + k0 + 8);
                gh[s][g] = __builtin_amdgcn_wmma_f32_16x16x32_bf16(
                    false, ah.v, false, bh.v, (short)0, gh[s][g], false, false);
            }
        }
    }

    // epilogue: GRU nonlinearity, write h_new
    const int mrow0 = mtile * 16 + ((lane & 16) ? 8 : 0);
    #pragma unroll
    for (int s = 0; s < 2; ++s) {
        const int j = jbase + s * 16 + l15;
        const float bir = b_ih[j],          bhr = b_hh[j];
        const float biz = b_ih[HH + j],     bhz = b_hh[HH + j];
        const float bin = b_ih[2 * HH + j], bhn = b_hh[2 * HH + j];
        #pragma unroll
        for (int d = 0; d < 8; ++d) {
            const size_t idx = (size_t)(mrow0 + d) * HH + j;
            const float r  = 1.f / (1.f + __expf(-(gi[s][0][d] + bir +
                                                   gh[s][0][d] + bhr)));
            const float z  = 1.f / (1.f + __expf(-(gi[s][1][d] + biz +
                                                   gh[s][1][d] + bhz)));
            const float nn = tanhf(gi[s][2][d] + bin +
                                   r * (gh[s][2][d] + bhn));
            const float hn = (1.f - z) * nn + z * h_prev[idx];
            h_next[idx]    = hn;
            h_next_bf[idx] = (__bf16)hn;
        }
    }
}

// ---------------------------------------------------------------- WMMA GEMM
// D[M,N] = act(A[M,K] (row stride lda) * W[N,K]^T + bias). Each wave does an
// MR x 64 strip; interior strips take a guard-free fast path (wave-uniform
// branch), only the ragged last vocab strip is guarded.
template <int MR>
__global__ __launch_bounds__(128) void
wmma_gemm(const __bf16* __restrict__ A, int lda,
          const __bf16* __restrict__ W,          // [N, K] row-major
          const float*  __restrict__ bias,       // [N]
          float*  __restrict__ Cf, int ldc,      // optional f32 out
          __bf16* __restrict__ Cb, int ldcb,     // optional bf16 out
          int M, int N, int K, int relu)
{
    constexpr int MT = MR / 16;
    const int wave    = blockIdx.x * (blockDim.x >> 5) + (threadIdx.x >> 5);
    const int mtiles  = M / MR;
    const int nstrips = (N + 63) >> 6;
    if (wave >= mtiles * nstrips) return;
    const int mtile = wave % mtiles;
    const int nbase = (wave / mtiles) << 6;

    const int lane = threadIdx.x & 31;
    const int l15  = lane & 15;
    const int akb  = (lane & 16) ? 16 : 0;
    const int bkb  = (lane & 16) ? 16 : 0;
    const int bcol0 = nbase + l15;

    const __bf16* aptr[MT];
    #pragma unroll
    for (int i = 0; i < MT; ++i)
        aptr[i] = A + (size_t)(mtile * MR + i * 16 + l15) * lda + akb;

    v8f acc[MT][4];
    #pragma unroll
    for (int i = 0; i < MT; ++i)
        #pragma unroll
        for (int s = 0; s < 4; ++s) acc[i][s] = (v8f){};

    if (nbase + 64 <= N) {
        const __bf16* wptr[4];
        #pragma unroll
        for (int s = 0; s < 4; ++s)
            wptr[s] = W + (size_t)(bcol0 + s * 16) * K + bkb;

        for (int k0 = 0; k0 < K; k0 += 32) {
            ABFrag af[MT];
            #pragma unroll
            for (int i = 0; i < MT; ++i) {
                af[i].h[0] = *reinterpret_cast<const v8bf*>(aptr[i] + k0);
                af[i].h[1] = *reinterpret_cast<const v8bf*>(aptr[i] + k0 + 8);
            }
            #pragma unroll
            for (int s = 0; s < 4; ++s) {
                ABFrag bf;
                bf.h[0] = *reinterpret_cast<const v8bf*>(wptr[s] + k0);
                bf.h[1] = *reinterpret_cast<const v8bf*>(wptr[s] + k0 + 8);
                #pragma unroll
                for (int i = 0; i < MT; ++i)
                    acc[i][s] = __builtin_amdgcn_wmma_f32_16x16x32_bf16(
                        false, af[i].v, false, bf.v, (short)0, acc[i][s],
                        false, false);
            }
        }
    } else {
        for (int k0 = 0; k0 < K; k0 += 32) {
            ABFrag af[MT];
            #pragma unroll
            for (int i = 0; i < MT; ++i) {
                af[i].h[0] = *reinterpret_cast<const v8bf*>(aptr[i] + k0);
                af[i].h[1] = *reinterpret_cast<const v8bf*>(aptr[i] + k0 + 8);
            }
            #pragma unroll
            for (int s = 0; s < 4; ++s) {
                const int bcol = bcol0 + (s << 4);
                ABFrag bf = {};
                if (bcol < N) {
                    const __bf16* wp = W + (size_t)bcol * K + k0 + bkb;
                    bf.h[0] = *reinterpret_cast<const v8bf*>(wp);
                    bf.h[1] = *reinterpret_cast<const v8bf*>(wp + 8);
                }
                #pragma unroll
                for (int i = 0; i < MT; ++i)
                    acc[i][s] = __builtin_amdgcn_wmma_f32_16x16x32_bf16(
                        false, af[i].v, false, bf.v, (short)0, acc[i][s],
                        false, false);
            }
        }
    }

    #pragma unroll
    for (int i = 0; i < MT; ++i) {
        const int mrow0 = mtile * MR + i * 16 + ((lane & 16) ? 8 : 0);
        #pragma unroll
        for (int s = 0; s < 4; ++s) {
            const int n = bcol0 + (s << 4);
            if (n >= N) continue;
            const float bv = bias ? bias[n] : 0.f;
            #pragma unroll
            for (int d = 0; d < 8; ++d) {
                const int m = mrow0 + d;
                float v = acc[i][s][d] + bv;
                if (relu) v = fmaxf(v, 0.f);
                if (Cf) Cf[(size_t)m * ldc  + n] = v;
                if (Cb) Cb[(size_t)m * ldcb + n] = (__bf16)v;
            }
        }
    }
}

// ---------------------------------------------------------------- host side
extern "C" void kernel_launch(void* const* d_in, const int* in_sizes, int n_in,
                              void* d_out, int out_size, void* d_ws, size_t ws_size,
                              hipStream_t stream) {
    (void)in_sizes; (void)n_in; (void)out_size; (void)ws_size;

    const float* C           = (const float*)d_in[0];   // [B]
    const float* init_hidden = (const float*)d_in[1];   // [B,H]
    const float* W_ih        = (const float*)d_in[2];   // [3H,I]
    const float* W_hh        = (const float*)d_in[3];   // [3H,H]
    const float* b_ih        = (const float*)d_in[4];   // [3H]
    const float* b_hh        = (const float*)d_in[5];   // [3H]
    const float* linC_w      = (const float*)d_in[6];   // [I,1]
    const float* linC_b      = (const float*)d_in[7];   // [I]
    const float* lin_w       = (const float*)d_in[8];   // [I,H]
    const float* lin_b       = (const float*)d_in[9];   // [I]
    const float* linres_w    = (const float*)d_in[10];  // [V,I]
    const float* linres_b    = (const float*)d_in[11];  // [V]
    float* logits = (float*)d_out;                      // [B,T,V]

    // workspace carve-up (256B aligned)
    char* ws = (char*)d_ws;
    size_t off = 0;
    auto carve = [&](size_t bytes) {
        void* p = ws + off;
        off = (off + bytes + 255) & ~(size_t)255;
        return p;
    };
    __bf16* W_ih_bf   = (__bf16*)carve((size_t)3 * HH * II * 2);
    __bf16* W_hh_bf   = (__bf16*)carve((size_t)3 * HH * HH * 2);
    __bf16* lin_w_bf  = (__bf16*)carve((size_t)II * HH * 2);
    __bf16* linres_bf = (__bf16*)carve((size_t)VV * II * 2);
    __bf16* x0_bf     = (__bf16*)carve((size_t)BB * II * 2);
    float*  h_f32[2]  = { (float*)carve((size_t)BB * HH * 4),
                          (float*)carve((size_t)BB * HH * 4) };
    __bf16* h_bf[2]   = { (__bf16*)carve((size_t)BB * HH * 2),
                          (__bf16*)carve((size_t)BB * HH * 2) };
    __bf16* outs_bf   = (__bf16*)carve((size_t)BB * TT * II * 2);  // [B*T, I]

    // 1) weights -> bf16 (linres_w shrinks 206MB->103MB: becomes L2-resident)
    cvt_f32_bf16<<<4096, 256, 0, stream>>>(W_ih,     W_ih_bf,   (long long)3 * HH * II);
    cvt_f32_bf16<<<4096, 256, 0, stream>>>(W_hh,     W_hh_bf,   (long long)3 * HH * HH);
    cvt_f32_bf16<<<2048, 256, 0, stream>>>(lin_w,    lin_w_bf,  (long long)II * HH);
    cvt_f32_bf16<<<8192, 256, 0, stream>>>(linres_w, linres_bf, (long long)VV * II);

    // 2) x0 and h0
    init_kernel<<<(BB * II) / 256, 256, 0, stream>>>(
        C, linC_w, linC_b, init_hidden, x0_bf, h_f32[0], h_bf[0]);

    auto gemm16 = [&](const __bf16* A, int lda, const __bf16* W, const float* bias,
                      float* Cf, int ldc, __bf16* Cb, int ldcb, int M, int N) {
        int waves  = (M / 16) * ((N + 63) >> 6);
        int blocks = (waves + 3) / 4;
        wmma_gemm<16><<<blocks, 128, 0, stream>>>(A, lda, W, bias, Cf, ldc,
                                                  Cb, ldcb, M, N, II, 0);
    };

    // 3) 30 sequential GRU steps: 2 dependency levels per step
    //    (fused gates+nonlinearity, then the output projection)
    for (int t = 0; t < TT; ++t) {
        const __bf16* x   = (t == 0) ? x0_bf : (outs_bf + (size_t)(t - 1) * II);
        const int     xld = (t == 0) ? II : TT * II;
        const int cur = t & 1, nxt = cur ^ 1;
        gru_step_fused<<<64, 128, 0, stream>>>(          // 256 waves
            x, xld, W_ih_bf, W_hh_bf, b_ih, b_hh,
            h_f32[cur], h_bf[cur], h_f32[nxt], h_bf[nxt]);
        // out_t = h_new @ lin_w^T + lin_b -> bf16 row (b*T + t) of outs matrix
        gemm16(h_bf[nxt], II, lin_w_bf, lin_b,
               nullptr, 0, outs_bf + (size_t)t * II, TT * II, BB, II);
    }

    // 4) logits = relu(outs @ linres_w^T + linres_b)  [3840 x 50257]
    //    HBM-store bound (772MB out @ 23.3TB/s ~= 33us); weights stream from L2.
    {
        int waves  = ((BB * TT) / 32) * ((VV + 63) >> 6);
        int blocks = (waves + 3) / 4;
        wmma_gemm<32><<<blocks, 128, 0, stream>>>(
            outs_bf, II, linres_bf, linres_b,
            logits, VV, nullptr, 0, BB * TT, VV, II, 1);
    }
}
